// Block_84258668413379
// MI455X (gfx1250) — matrix operand order repaired and checked by reading, not compiled, not run on hardware
//
#include <hip/hip_runtime.h>
#include <hip/hip_bf16.h>
#include <math.h>

// ---------------- problem constants ----------------
#define BB   16
#define NN   512
#define CC   768
#define HH   12
#define DD   64          // CC/HH
#define SPLIT 256
#define TABLE 1024
#define HID  3072        // 4*CC
#define MM   (BB*NN)     // 8192 tokens
#define SCALE 0.125f     // D^-0.5 = 64^-0.5

typedef __bf16 v16bf __attribute__((ext_vector_type(16)));
typedef float  v8f   __attribute__((ext_vector_type(8)));
typedef unsigned int uint4v __attribute__((ext_vector_type(4)));

union FragU { v16bf f; uint4v u2[2]; };

// A-matrix 16x32 bf16 fragment (ISA layout): lane half=lane>>4 selects K sub-block.
// lane holds K in {kb+half*8 .. +7} U {kb+16+half*8 .. +7}, row M = lane&15.
static __device__ inline v16bf load_frag_a(const __bf16* rowPtr, int kb, int half) {
    FragU fr;
    fr.u2[0] = *reinterpret_cast<const uint4v*>(rowPtr + kb + half * 8);
    fr.u2[1] = *reinterpret_cast<const uint4v*>(rowPtr + kb + 16 + half * 8);
    return fr.f;
}
// B-matrix 32x16 bf16 fragment: lane n = lane&15 is the output column; lane holds
// K in {kb+half*8 .. +7} U {kb+16+half*8 .. +7} along the weight row (W stored [Nout,K]).
static __device__ inline v16bf load_frag_b(const __bf16* colPtr, int kb, int half) {
    FragU fr;
    int k0 = kb + half * 8;
    fr.u2[0] = *reinterpret_cast<const uint4v*>(colPtr + k0);
    fr.u2[1] = *reinterpret_cast<const uint4v*>(colPtr + k0 + 16);
    return fr.f;
}

static __device__ inline float gelu_exact(float v) {
    return 0.5f * v * (1.0f + erff(v * 0.70710678118654752f));
}

// ---------------- gfx1250 async global->LDS copy (ASYNCcnt path) ----------------
#if defined(__has_builtin)
#  if __has_builtin(__builtin_amdgcn_global_load_async_to_lds_b128)
#    define HAVE_ASYNC_LDS 1
#  endif
#endif

#ifdef HAVE_ASYNC_LDS
// builtin signature (from compiler diagnostic): (v4i AS1*, v4i AS3*, imm int, imm int)
typedef int v4i __attribute__((vector_size(16)));
typedef __attribute__((address_space(1))) v4i* gbl_v4i_ptr;
typedef __attribute__((address_space(3))) v4i* lds_v4i_ptr;
#endif

static __device__ inline void copy16_to_lds(__bf16* l, const __bf16* g) {
#ifdef HAVE_ASYNC_LDS
    __builtin_amdgcn_global_load_async_to_lds_b128(
        (gbl_v4i_ptr)(g), (lds_v4i_ptr)(l), 0, 0);
    __builtin_amdgcn_global_load_async_to_lds_b128(
        (gbl_v4i_ptr)(g + 8), (lds_v4i_ptr)(l + 8), 0, 0);
#else
    *reinterpret_cast<uint4v*>(l)     = *reinterpret_cast<const uint4v*>(g);
    *reinterpret_cast<uint4v*>(l + 8) = *reinterpret_cast<const uint4v*>(g + 8);
#endif
}

static __device__ inline void wait_async_lds() {
#ifdef HAVE_ASYNC_LDS
#  if __has_builtin(__builtin_amdgcn_s_wait_asynccnt)
    __builtin_amdgcn_s_wait_asynccnt(0);
#  else
    asm volatile("s_wait_asynccnt 0" ::: "memory");
#  endif
#endif
}

// ---------------- f32 -> bf16 convert ----------------
__global__ __launch_bounds__(256) void cvt_f32_bf16(const float* __restrict__ src,
                                                    __bf16* __restrict__ dst, int n) {
    int i = blockIdx.x * 256 + threadIdx.x;
    if (i < n) dst[i] = (__bf16)src[i];
}

// ---------------- LayerNorm (split-aware) -> bf16 ----------------
__global__ __launch_bounds__(256) void ln_kernel(const float* __restrict__ x,
                                                 const float* __restrict__ gT, const float* __restrict__ bT,
                                                 const float* __restrict__ gF, const float* __restrict__ bF,
                                                 __bf16* __restrict__ out, int splitTok) {
    int m = blockIdx.x;                 // token 0..MM-1
    int n = m % NN;
    const float* g = (n < splitTok) ? gT : gF;
    const float* b = (n < splitTok) ? bT : bF;
    const float* row = x + (size_t)m * CC;

    __shared__ float s1[256], s2[256];
    float s = 0.f, ss = 0.f;
    for (int c = threadIdx.x; c < CC; c += 256) {
        float v = row[c]; s += v; ss += v * v;
    }
    s1[threadIdx.x] = s; s2[threadIdx.x] = ss;
    __syncthreads();
    for (int st = 128; st > 0; st >>= 1) {
        if (threadIdx.x < st) { s1[threadIdx.x] += s1[threadIdx.x + st]; s2[threadIdx.x] += s2[threadIdx.x + st]; }
        __syncthreads();
    }
    float mean = s1[0] * (1.0f / CC);
    float var  = s2[0] * (1.0f / CC) - mean * mean;
    float inv  = rsqrtf(var + 1e-5f);
    for (int c = threadIdx.x; c < CC; c += 256) {
        out[(size_t)m * CC + c] = (__bf16)((row[c] - mean) * inv * g[c] + b[c]);
    }
}

// ---------------- tiled WMMA GEMM ----------------
// Workgroup: 256 threads = 8 waves (2 M x 4 N). WG tile: 128x128, BK=32.
// Each wave: 64x32 register block = 4x2 16x16 accumulators, 8 WMMAs / K-step.
// LDS: double-buffered A/B tiles, rows padded to 40 elems (80 B, 16B-aligned rows).
// out[m, col] = epilogue( sum_k A[m,k] * W[col,k] )
// mode 0: bf16 out  mode 1: gelu -> bf16  mode 2: resid + gamma*(acc+bias) -> f32
#define LDS_STRIDE 40
__global__ __launch_bounds__(256) void wmma_gemm_tiled(
    const __bf16* __restrict__ A, int lda,
    const __bf16* __restrict__ Wt, const __bf16* __restrict__ Wf, int ldw,
    const float* __restrict__ biasT, const float* __restrict__ biasF,
    const float* __restrict__ gamma, const float* __restrict__ resid,
    float* __restrict__ outF, __bf16* __restrict__ outB, int ldo,
    int K, int mode, int splitTok) {

    __shared__ __align__(16) __bf16 As[2][128][LDS_STRIDE];
    __shared__ __align__(16) __bf16 Bs[2][128][LDS_STRIDE];

    int tileM = blockIdx.x, tileN = blockIdx.y;
    int tid   = threadIdx.x;
    int wave  = tid >> 5;
    int lane  = tid & 31;
    int half  = lane >> 4;
    int mn    = lane & 15;
    int waveM = wave >> 2;      // 0..1
    int waveN = wave & 3;       // 0..3

    bool useT = ((tileM * 128) % NN) < splitTok;
    const __bf16* W    = useT ? Wt : Wf;
    const float*  bias = useT ? biasT : biasF;

    // cooperative tile load assignment: each thread stages 16 bf16 of A and of B
    int ldRow = tid >> 1;                 // 0..127
    int ldCol = (tid & 1) * 16;           // 0 or 16
    const __bf16* gA = A + (size_t)(tileM * 128 + ldRow) * lda + ldCol;
    const __bf16* gW = W + (size_t)(tileN * 128 + ldRow) * ldw + ldCol;

    v8f acc[4][2];
#pragma unroll
    for (int mt = 0; mt < 4; mt++)
#pragma unroll
        for (int nt = 0; nt < 2; nt++)
            acc[mt][nt] = v8f{0.f, 0.f, 0.f, 0.f, 0.f, 0.f, 0.f, 0.f};

    // prologue: stage first K-slab into buffer 0
    copy16_to_lds(&As[0][ldRow][ldCol], gA);
    copy16_to_lds(&Bs[0][ldRow][ldCol], gW);

    int buf = 0;
    for (int kb = 0; kb < K; kb += 32) {
        wait_async_lds();
        __syncthreads();                       // slab `buf` visible to all waves

        if (kb + 32 < K) {                     // prefetch next slab into other buffer
            copy16_to_lds(&As[buf ^ 1][ldRow][ldCol], gA + kb + 32);
            copy16_to_lds(&Bs[buf ^ 1][ldRow][ldCol], gW + kb + 32);
        }

        v16bf bfrag[2];
#pragma unroll
        for (int nt = 0; nt < 2; nt++)
            bfrag[nt] = load_frag_b(&Bs[buf][waveN * 32 + nt * 16 + mn][0], 0, half);
#pragma unroll
        for (int mt = 0; mt < 4; mt++) {
            v16bf afrag = load_frag_a(&As[buf][waveM * 64 + mt * 16 + mn][0], 0, half);
#pragma unroll
            for (int nt = 0; nt < 2; nt++)
                acc[mt][nt] = __builtin_amdgcn_wmma_f32_16x16x32_bf16(
                    false, afrag, false, bfrag[nt], (short)0, acc[mt][nt], false, false);
        }
        __syncthreads();                       // done reading `buf` before it is re-filled
        buf ^= 1;
    }

    // epilogue
#pragma unroll
    for (int nt = 0; nt < 2; nt++) {
        int col = tileN * 128 + waveN * 32 + nt * 16 + mn;
        float bv = bias  ? bias[col]  : 0.f;
        float gv = gamma ? gamma[col] : 0.f;
#pragma unroll
        for (int mt = 0; mt < 4; mt++) {
#pragma unroll
            for (int r = 0; r < 8; r++) {
                int m = tileM * 128 + waveM * 64 + mt * 16 + r + half * 8;
                float v = acc[mt][nt][r] + bv;
                if (mode == 1) v = gelu_exact(v);
                if (mode == 2) v = resid[(size_t)m * ldo + col] + gv * v;
                if (outF) outF[(size_t)m * ldo + col] = v;
                else      outB[(size_t)m * ldo + col] = (__bf16)v;
            }
        }
    }
}

// ---------------- split qkv -> q(scaled) [B,H,N,D], k [B,H,N,D], vT [B,H,D,N] ----------------
__global__ __launch_bounds__(256) void split_qkv_kernel(const __bf16* __restrict__ qkv,
                                                        __bf16* __restrict__ q,
                                                        __bf16* __restrict__ k,
                                                        __bf16* __restrict__ vT) {
    int idx = blockIdx.x * 256 + threadIdx.x;            // over B*H*N*D
    int d = idx % DD;
    int n = (idx / DD) % NN;
    int h = (idx / (DD * NN)) % HH;
    int b = idx / (DD * NN * HH);
    size_t row = ((size_t)(b * NN + n)) * (3 * CC);
    int c = h * DD + d;
    q [((size_t)((b * HH + h) * NN) + n) * DD + d] = (__bf16)((float)qkv[row + c] * SCALE);
    k [((size_t)((b * HH + h) * NN) + n) * DD + d] = qkv[row + CC + c];
    vT[((size_t)((b * HH + h) * DD) + d) * NN + n] = qkv[row + 2 * CC + c];
}

// ---------------- attention: one WG per (b, h, 16-row tile) ----------------
__global__ __launch_bounds__(256) void attn_kernel(const __bf16* __restrict__ q,
                                                   const __bf16* __restrict__ kbuf,
                                                   const __bf16* __restrict__ vT,
                                                   const float* __restrict__ rel_table,
                                                   const int* __restrict__ rpi,
                                                   const unsigned char* __restrict__ mask,
                                                   __bf16* __restrict__ o) {
    int rowTile = blockIdx.x;       // 0..31
    int h = blockIdx.y;             // 0..11
    int b = blockIdx.z;             // 0..15

    __shared__ __align__(16) float  S [16][NN];   // 32 KB scores
    __shared__ __align__(16) __bf16 Pb[16][NN];   // 16 KB probabilities (bf16)
    __shared__ float red[16][16];

    int tid  = threadIdx.x;
    int wave = tid >> 5;
    int lane = tid & 31;
    int half = lane >> 4;
    int mn   = lane & 15;

    size_t headBase = (size_t)(b * HH + h) * NN;       // rows of q/k
    const float NEG_INF = -__builtin_inff();

    // ---- phase A: S = q * k^T  (+bias, mask) ----
    const __bf16* qrow = q + (headBase + rowTile * 16 + mn) * DD;
    for (int ct = wave; ct < NN / 16; ct += 8) {
        const __bf16* krow = kbuf + (headBase + ct * 16 + mn) * DD;
        v8f acc = {0.f, 0.f, 0.f, 0.f, 0.f, 0.f, 0.f, 0.f};
#pragma unroll
        for (int kb = 0; kb < DD; kb += 32) {
            v16bf a = load_frag_a(qrow, kb, half);
            v16bf bfr = load_frag_b(krow, kb, half);
            acc = __builtin_amdgcn_wmma_f32_16x16x32_bf16(false, a, false, bfr, (short)0, acc, false, false);
        }
        int j = ct * 16 + mn;
        bool mok = mask[b * NN + j] != 0;
#pragma unroll
        for (int r = 0; r < 8; r++) {
            int iLoc = r + half * 8;
            int iGlb = rowTile * 16 + iLoc;
            int idx  = rpi[(size_t)iGlb * NN + j];
            float v  = acc[r] + rel_table[(size_t)idx * HH + h];
            S[iLoc][j] = mok ? v : NEG_INF;
        }
    }
    __syncthreads();

    // ---- phase B: softmax over 512 columns, 16 threads per row ----
    int row = tid >> 4;
    int t   = tid & 15;
    float mx = NEG_INF;
    for (int c = t; c < NN; c += 16) mx = fmaxf(mx, S[row][c]);
    red[row][t] = mx;
    __syncthreads();
    for (int st = 8; st > 0; st >>= 1) {
        if (t < st) red[row][t] = fmaxf(red[row][t], red[row][t + st]);
        __syncthreads();
    }
    float rowmax = red[row][0];
    __syncthreads();
    float sum = 0.f;
    for (int c = t; c < NN; c += 16) {
        float e = __expf(S[row][c] - rowmax);
        S[row][c] = e;
        sum += e;
    }
    red[row][t] = sum;
    __syncthreads();
    for (int st = 8; st > 0; st >>= 1) {
        if (t < st) red[row][t] += red[row][t + st];
        __syncthreads();
    }
    float inv = 1.0f / red[row][0];
    for (int c = t; c < NN; c += 16) Pb[row][c] = (__bf16)(S[row][c] * inv);
    __syncthreads();

    // ---- phase C: O = P * V  (waves 0..3 each own one 16-wide d tile) ----
    if (wave < DD / 16) {
        int dt = wave;
        const __bf16* prow = &Pb[mn][0];
        const __bf16* vrow = vT + ((size_t)(b * HH + h) * DD + dt * 16 + mn) * NN;
        v8f acc = {0.f, 0.f, 0.f, 0.f, 0.f, 0.f, 0.f, 0.f};
        for (int kb = 0; kb < NN; kb += 32) {
            v16bf a = load_frag_a(prow, kb, half);       // ds_load_b128 from LDS
            v16bf bfr = load_frag_b(vrow, kb, half);     // contiguous V^T row
            acc = __builtin_amdgcn_wmma_f32_16x16x32_bf16(false, a, false, bfr, (short)0, acc, false, false);
        }
#pragma unroll
        for (int r = 0; r < 8; r++) {
            int n = rowTile * 16 + r + half * 8;
            o[((size_t)(b * NN + n)) * CC + h * DD + dt * 16 + mn] = (__bf16)acc[r];
        }
    }
}

// ---------------- host-side launch ----------------
static inline size_t alignUp(size_t v) { return (v + 255) & ~(size_t)255; }

extern "C" void kernel_launch(void* const* d_in, const int* in_sizes, int n_in,
                              void* d_out, int out_size, void* d_ws, size_t ws_size,
                              hipStream_t stream) {
    (void)in_sizes; (void)n_in; (void)out_size; (void)ws_size;

    const float* x        = (const float*)d_in[0];
    const unsigned char* mask = (const unsigned char*)d_in[1];
    const int*   rpi      = (const int*)d_in[2];
    const float* W_qkv    = (const float*)d_in[3];
    const float* W_proj   = (const float*)d_in[4];
    const float* b_proj   = (const float*)d_in[5];
    const float* rel_tab  = (const float*)d_in[6];
    const float* ln1_g    = (const float*)d_in[7];
    const float* ln1_b    = (const float*)d_in[8];
    const float* gamma1   = (const float*)d_in[9];
    const float* gamma2   = (const float*)d_in[10];
    const float* ln2t_g   = (const float*)d_in[11];
    const float* ln2t_b   = (const float*)d_in[12];
    const float* fc1t_W   = (const float*)d_in[13];
    const float* fc1t_b   = (const float*)d_in[14];
    const float* fc2t_W   = (const float*)d_in[15];
    const float* fc2t_b   = (const float*)d_in[16];
    const float* ln2f_g   = (const float*)d_in[17];
    const float* ln2f_b   = (const float*)d_in[18];
    const float* fc1f_W   = (const float*)d_in[19];
    const float* fc1f_b   = (const float*)d_in[20];
    const float* fc2f_W   = (const float*)d_in[21];
    const float* fc2f_b   = (const float*)d_in[22];
    float* out = (float*)d_out;

    // ---- workspace carve-up ----
    char* w = (char*)d_ws;
    size_t off = 0;
    auto take = [&](size_t bytes) { char* p = w + off; off += alignUp(bytes); return p; };

    __bf16* wqkv_bf  = (__bf16*)take((size_t)3 * CC * CC * 2);
    __bf16* wproj_bf = (__bf16*)take((size_t)CC * CC * 2);
    __bf16* w1t_bf   = (__bf16*)take((size_t)HID * CC * 2);
    __bf16* w2t_bf   = (__bf16*)take((size_t)CC * HID * 2);
    __bf16* w1f_bf   = (__bf16*)take((size_t)HID * CC * 2);
    __bf16* w2f_bf   = (__bf16*)take((size_t)CC * HID * 2);
    __bf16* h_bf     = (__bf16*)take((size_t)MM * CC * 2);          // LN1 out; reused for LN2 out
    __bf16* qkv_bf   = (__bf16*)take((size_t)MM * 3 * CC * 2);      // aliased later by h1
    __bf16* q_bf     = (__bf16*)take((size_t)BB * HH * NN * DD * 2);
    __bf16* k_bf     = (__bf16*)take((size_t)BB * HH * NN * DD * 2);
    __bf16* vT_bf    = (__bf16*)take((size_t)BB * HH * DD * NN * 2);
    __bf16* o_bf     = (__bf16*)take((size_t)MM * CC * 2);
    float*  x1       = (float*)take((size_t)MM * CC * 4);
    __bf16* h1_bf    = qkv_bf;   // alias: qkv (37.7MB) + q (12.6MB) == h1 (50.3MB); both dead by fc1

    // ---- 1) weights f32 -> bf16 ----
    auto cvt = [&](const float* s, __bf16* d, int n) {
        cvt_f32_bf16<<<(n + 255) / 256, 256, 0, stream>>>(s, d, n);
    };
    cvt(W_qkv,  wqkv_bf,  3 * CC * CC);
    cvt(W_proj, wproj_bf, CC * CC);
    cvt(fc1t_W, w1t_bf,   HID * CC);
    cvt(fc2t_W, w2t_bf,   CC * HID);
    cvt(fc1f_W, w1f_bf,   HID * CC);
    cvt(fc2f_W, w2f_bf,   CC * HID);

    // ---- 2) LN1 ----
    ln_kernel<<<MM, 256, 0, stream>>>(x, ln1_g, ln1_b, ln1_g, ln1_b, h_bf, NN);

    // ---- 3) QKV GEMM (8192 x 2304 x 768) ----
    wmma_gemm_tiled<<<dim3(MM / 128, 3 * CC / 128), 256, 0, stream>>>(
        h_bf, CC, wqkv_bf, wqkv_bf, CC, nullptr, nullptr, nullptr, nullptr,
        nullptr, qkv_bf, 3 * CC, CC, 0, NN);

    // ---- 4) split qkv ----
    split_qkv_kernel<<<(BB * HH * NN * DD) / 256, 256, 0, stream>>>(qkv_bf, q_bf, k_bf, vT_bf);

    // ---- 5) attention ----
    attn_kernel<<<dim3(NN / 16, HH, BB), 256, 0, stream>>>(q_bf, k_bf, vT_bf, rel_tab, rpi, mask, o_bf);

    // ---- 6) proj + residual (gamma1) -> x1 f32 ----
    wmma_gemm_tiled<<<dim3(MM / 128, CC / 128), 256, 0, stream>>>(
        o_bf, CC, wproj_bf, wproj_bf, CC, b_proj, b_proj, gamma1, x,
        x1, nullptr, CC, CC, 2, NN);

    // ---- 7) LN2 (split-aware) ----
    ln_kernel<<<MM, 256, 0, stream>>>(x1, ln2t_g, ln2t_b, ln2f_g, ln2f_b, h_bf, SPLIT);

    // ---- 8) fc1 + GELU (split weights) ----
    wmma_gemm_tiled<<<dim3(MM / 128, HID / 128), 256, 0, stream>>>(
        h_bf, CC, w1t_bf, w1f_bf, CC, fc1t_b, fc1f_b, nullptr, nullptr,
        nullptr, h1_bf, HID, CC, 1, SPLIT);

    // ---- 9) fc2 + residual (gamma2) -> out ----
    wmma_gemm_tiled<<<dim3(MM / 128, CC / 128), 256, 0, stream>>>(
        h1_bf, HID, w2t_bf, w2f_bf, HID, fc2t_b, fc2f_b, gamma2, x1,
        out, nullptr, CC, HID, 2, SPLIT);
}
